// BilinearHead_60584808677393
// MI455X (gfx1250) — matrix operand order repaired and checked by reading, not compiled, not run on hardware
//
#include <hip/hip_runtime.h>

// Problem sizes (B, C, W, H, D, V) = (1024, 256, 8, 8, 64, 1968), HW = 64
#define C_CH     256
#define HW       64
#define D_DIM    64
#define V_MOVES  1968
#define XPITCH   272   // ushort elems per spatial row of bf16 activations (32B-aligned rows)
#define FTPITCH  68    // float elems per spatial row of f/t buffers (16B-aligned rows)

typedef __attribute__((ext_vector_type(16))) __bf16 v16bf;
typedef __attribute__((ext_vector_type(8)))  float  v8f;

static __device__ __forceinline__ unsigned short f2bf(float f) {
    unsigned int u = __float_as_uint(f);
    u += 0x7FFFu + ((u >> 16) & 1u);           // round-to-nearest-even
    return (unsigned short)(u >> 16);
}

// ---------------------------------------------------------------------------
// Pre-pack the two 64x256 f32 weights into bf16 A-fragment layout:
// flat index = (((which*4 + mt)*8 + ks)*32 + lane)*16 + j
//   lane 0-15 : M = mt*16 + lane,      K = ks*32 +      j
//   lane 16-31: M = mt*16 + (lane-16), K = ks*32 + 16 + j
// (contiguous k order, matching the B-fragment fill)
// ---------------------------------------------------------------------------
__global__ __launch_bounds__(256) void pack_weights_kernel(
    const float* __restrict__ from_w, const float* __restrict__ to_w,
    unsigned short* __restrict__ wpack)
{
    int idx = blockIdx.x * 256 + threadIdx.x;   // 0..2047
    if (idx >= 2048) return;
    int lane  = idx & 31;
    int ks    = (idx >> 5) & 7;
    int mt    = (idx >> 8) & 3;
    int which = idx >> 10;
    const float* W = which ? to_w : from_w;
    int m    = mt * 16 + (lane & 15);
    int koff = (lane >> 4) * 16;
    unsigned short* dst = wpack + (size_t)idx * 16;
#pragma unroll
    for (int j = 0; j < 16; ++j) {
        int k = ks * 32 + koff + j;
        dst[j] = f2bf(W[m * C_CH + k]);
    }
}

// ---------------------------------------------------------------------------
// Fused: RMSNorm stats -> bf16 staging -> 2x WMMA GEMM -> gather/bilinear dot
// One block (256 threads = 8 waves) per batch element.
// ---------------------------------------------------------------------------
__global__ __launch_bounds__(256) void bilinear_head_kernel(
    const float* __restrict__ x, const float* __restrict__ nw,
    const unsigned short* __restrict__ wpack,
    const float* __restrict__ from_b, const float* __restrict__ to_b,
    const float* __restrict__ promo_bias,
    const int* __restrict__ from_idx, const int* __restrict__ to_idx,
    const int* __restrict__ promo_idx,
    float* __restrict__ out)
{
    // xnbf (64 x XPITCH bf16 = 34816 B) aliases ftbuf (2 x 64 x FTPITCH f32 = 34816 B)
    __shared__ __align__(32) unsigned char smem[2 * HW * FTPITCH * 4 + 64];
    unsigned short* xnbf = (unsigned short*)smem;          // [s][c] transposed, bf16
    float* ftbuf = (float*)smem;                           // [2][s][d], f32
    float* red   = (float*)(smem + 2 * HW * FTPITCH * 4);  // reduction scratch

    const int tid = threadIdx.x;
    const int b   = blockIdx.x;

    // ---- Phase 1: sum(x^2) + store bf16(x * norm_weight) transposed to LDS ----
    // Thread tid owns channel c = tid (256 threads == 256 channels).
    const float* xc = x  + (size_t)b * (C_CH * HW) + tid * HW;
    const float* nc = nw + tid * HW;
    float ss = 0.0f;
#pragma unroll
    for (int s = 0; s < HW; s += 4) {
        float4 xv = *(const float4*)(xc + s);
        float4 wv = *(const float4*)(nc + s);
        ss += xv.x * xv.x + xv.y * xv.y + xv.z * xv.z + xv.w * xv.w;
        xnbf[(s + 0) * XPITCH + tid] = f2bf(xv.x * wv.x);
        xnbf[(s + 1) * XPITCH + tid] = f2bf(xv.y * wv.y);
        xnbf[(s + 2) * XPITCH + tid] = f2bf(xv.z * wv.z);
        xnbf[(s + 3) * XPITCH + tid] = f2bf(xv.w * wv.w);
    }
#pragma unroll
    for (int off = 16; off > 0; off >>= 1) ss += __shfl_down(ss, off, 32);
    if ((tid & 31) == 0) red[tid >> 5] = ss;
    __syncthreads();
    if (tid == 0) {
        float tot = 0.f;
#pragma unroll
        for (int i = 0; i < 8; ++i) tot += red[i];
        red[8] = 1.0f / sqrtf(tot * (1.0f / 16384.0f) + 1e-6f);
    }
    __syncthreads();
    const float inv_rms = red[8];

    // ---- Phase 2: WMMA GEMM.  Wave wv -> (which = f/t, mt = 16-row tile),
    //      4 accumulators cover all 64 output columns; K loop: 8 x 32. ----
    const int lane  = tid & 31;
    const int wv    = tid >> 5;
    const int which = wv >> 2;          // 0: from_w, 1: to_w
    const int mt    = wv & 3;
    const int nlo   = lane & 15;
    const int khalf = (lane >> 4) * 16; // B-fragment: lanes 16-31 hold K+16..K+31
    const unsigned short* wp = wpack + (size_t)((which * 4 + mt) * 8) * 32 * 16;

    v8f acc[4] = {};
#pragma unroll
    for (int ks = 0; ks < 8; ++ks) {
        v16bf a = *(const v16bf*)(wp + (ks * 32 + lane) * 16);      // global, 32B/lane
        const unsigned short* bp = xnbf + ks * 32 + khalf;
#pragma unroll
        for (int nt = 0; nt < 4; ++nt) {
            v16bf bf = *(const v16bf*)(bp + (nt * 16 + nlo) * XPITCH); // LDS, 32B/lane
            acc[nt] = __builtin_amdgcn_wmma_f32_16x16x32_bf16(
                false, a, false, bf, (short)0, acc[nt], false, false);
        }
    }

    __syncthreads();   // all B-fragment reads done before ftbuf overwrites xnbf

    // ---- Epilogue: D = inv_rms * acc + bias, stored transposed as [s][d] ----
    {
        const float* bias = which ? to_b : from_b;
        float* dst = ftbuf + which * HW * FTPITCH;
#pragma unroll
        for (int r = 0; r < 8; ++r) {
            int d = mt * 16 + (lane >> 4) * 8 + r;   // C/D layout: M = r + 8*(lane>=16)
            float bb = bias[d];
#pragma unroll
            for (int nt = 0; nt < 4; ++nt) {
                int s = nt * 16 + nlo;
                dst[s * FTPITCH + d] = acc[nt][r] * inv_rms + bb;
            }
        }
    }
    __syncthreads();

    // ---- Phase 3: score[v] = dot_d(f[:,from_idx], t[:,to_idx]) + promo ----
    const float* fb = ftbuf;
    const float* tb = ftbuf + HW * FTPITCH;
    for (int v = tid; v < V_MOVES; v += 256) {
        const float* fr = fb + from_idx[v] * FTPITCH;
        const float* tr = tb + to_idx[v]   * FTPITCH;
        float sum = 0.f;
#pragma unroll
        for (int d = 0; d < D_DIM; d += 4) {
            float4 a4 = *(const float4*)(fr + d);
            float4 b4 = *(const float4*)(tr + d);
            sum += a4.x * b4.x + a4.y * b4.y + a4.z * b4.z + a4.w * b4.w;
        }
        out[(size_t)b * V_MOVES + v] = sum + promo_bias[promo_idx[v]];
    }
}

// ---------------------------------------------------------------------------
extern "C" void kernel_launch(void* const* d_in, const int* in_sizes, int n_in,
                              void* d_out, int out_size, void* d_ws, size_t ws_size,
                              hipStream_t stream) {
    (void)in_sizes; (void)n_in; (void)out_size; (void)ws_size;
    const float* x          = (const float*)d_in[0];
    const float* norm_w     = (const float*)d_in[1];
    const float* from_w     = (const float*)d_in[2];
    const float* from_b     = (const float*)d_in[3];
    const float* to_w       = (const float*)d_in[4];
    const float* to_b       = (const float*)d_in[5];
    const float* promo_bias = (const float*)d_in[6];
    const int*   from_idx   = (const int*)d_in[7];
    const int*   to_idx     = (const int*)d_in[8];
    const int*   promo_idx  = (const int*)d_in[9];
    float*       out        = (float*)d_out;
    unsigned short* wpack   = (unsigned short*)d_ws;   // 65536 bytes used

    pack_weights_kernel<<<8, 256, 0, stream>>>(from_w, to_w, wpack);
    bilinear_head_kernel<<<1024, 256, 0, stream>>>(
        x, norm_w, wpack, from_b, to_b, promo_bias,
        from_idx, to_idx, promo_idx, out);
}
